// DebugQuantizedLinear_64330020159895
// MI455X (gfx1250) — matrix operand order repaired and checked by reading, not compiled, not run on hardware
//
#include <hip/hip_runtime.h>

typedef __attribute__((ext_vector_type(16))) _Float16 v16h;
typedef __attribute__((ext_vector_type(8)))  float    v8f;

#define K_DIM 11008
#define N_DIM 4096
#define B_DIM 8192
#define NGRP  64     // N/64 quant groups

#define BM 128       // batch tile
#define BK 128       // output-K tile
#define SD 64        // stage depth along N (2 WMMA k-chunks; == one quant group)
#define LDH 72       // padded LDS row stride in halves (144B = 36 banks, conflict-free)

// Register-resident staging for one 64-deep stage: each thread owns
// 2 rows x 16 cols of both x and W_q, plus the shared mu1 chunk and
// per-row zero/scale (pre-multiplied by mu2).
struct Stage {
    float4 xa[2][4];
    int4   wa[2][4];
    float4 m1[4];
    float  z[2];
    float  s[2];
};

__device__ __forceinline__ void load_stage(const float* xbase, const int* wbase,
                                           const float* mu1c,
                                           const float* zrow0, const float* zrow1,
                                           const float* srow0, const float* srow1,
                                           float m2a, float m2b, int k0, Stage& st)
{
    #pragma unroll
    for (int c = 0; c < 4; ++c) {
        st.xa[0][c] = *(const float4*)(xbase + k0 + 4 * c);
        st.xa[1][c] = *(const float4*)(xbase + (size_t)64 * N_DIM + k0 + 4 * c);
        st.wa[0][c] = *(const int4*)  (wbase + k0 + 4 * c);
        st.wa[1][c] = *(const int4*)  (wbase + (size_t)64 * N_DIM + k0 + 4 * c);
        st.m1[c]    = *(const float4*)(mu1c + k0 + 4 * c);
    }
    const int g = k0 >> 6;            // stage is 64-aligned -> single quant group
    st.z[0] = zrow0[g];
    st.z[1] = zrow1[g];
    st.s[0] = srow0[g] * m2a;         // fold mu2[k] into the weight scale
    st.s[1] = srow1[g] * m2b;
}

__device__ __forceinline__ void deq_store(const Stage& st, _Float16* bufA, _Float16* bufW,
                                          int r, int colOff)
{
    #pragma unroll
    for (int rr = 0; rr < 2; ++rr) {
        alignas(16) _Float16 ya[16];
        alignas(16) _Float16 wv[16];
        const float z = st.z[rr];
        const float s = st.s[rr];
        #pragma unroll
        for (int c = 0; c < 4; ++c) {
            ya[4*c+0] = (_Float16)(st.xa[rr][c].x * st.m1[c].x);
            ya[4*c+1] = (_Float16)(st.xa[rr][c].y * st.m1[c].y);
            ya[4*c+2] = (_Float16)(st.xa[rr][c].z * st.m1[c].z);
            ya[4*c+3] = (_Float16)(st.xa[rr][c].w * st.m1[c].w);
            wv[4*c+0] = (_Float16)(((float)st.wa[rr][c].x - z) * s);
            wv[4*c+1] = (_Float16)(((float)st.wa[rr][c].y - z) * s);
            wv[4*c+2] = (_Float16)(((float)st.wa[rr][c].z - z) * s);
            wv[4*c+3] = (_Float16)(((float)st.wa[rr][c].w - z) * s);
        }
        _Float16* pA = bufA + (size_t)(r + rr * 64) * LDH + colOff;
        _Float16* pW = bufW + (size_t)(r + rr * 64) * LDH + colOff;
        *(uint4*)(pA)     = *(const uint4*)&ya[0];
        *(uint4*)(pA + 8) = *(const uint4*)&ya[8];
        *(uint4*)(pW)     = *(const uint4*)&wv[0];
        *(uint4*)(pW + 8) = *(const uint4*)&wv[8];
    }
}

__device__ __forceinline__ void compute_stage(const _Float16* bufA, const _Float16* bufW,
                                              int waveM, int waveK, int lane16, int khalf,
                                              v8f (&acc)[4][2])
{
    #pragma unroll
    for (int kc = 0; kc < 2; ++kc) {
        // A (16x32 f16): lane holds row m=lane16, one contiguous 16-half run
        // at n-offset khalf*16 (matches B's k order).
        v16h afrag[4];
        #pragma unroll
        for (int mt = 0; mt < 4; ++mt) {
            const _Float16* p = bufA + (size_t)(waveM * 64 + mt * 16 + lane16) * LDH
                                + kc * 32 + khalf * 16;
            ((uint4*)&afrag[mt])[0] = *(const uint4*)(p);
            ((uint4*)&afrag[mt])[1] = *(const uint4*)(p + 8);
        }
        // B (32x16 f16): lane holds col=lane16; lanes 0-15 Kred 0..15,
        // lanes 16-31 Kred 16..31 -> one contiguous 16-half run at khalf*16.
        v16h bfrag[2];
        #pragma unroll
        for (int kt = 0; kt < 2; ++kt) {
            const _Float16* p = bufW + (size_t)(waveK * 32 + kt * 16 + lane16) * LDH
                                + kc * 32 + khalf * 16;
            ((uint4*)&bfrag[kt])[0] = *(const uint4*)(p);
            ((uint4*)&bfrag[kt])[1] = *(const uint4*)(p + 8);
        }
        #pragma unroll
        for (int mt = 0; mt < 4; ++mt)
            #pragma unroll
            for (int kt = 0; kt < 2; ++kt)
                acc[mt][kt] = __builtin_amdgcn_wmma_f32_16x16x32_f16(
                    false, afrag[mt], false, bfrag[kt],
                    (short)0, acc[mt][kt], false, false);
    }
}

__global__ __launch_bounds__(256)
void qlinear_wmma_kernel(const float* __restrict__ x,
                         const int*   __restrict__ Wq,
                         const float* __restrict__ zeros,
                         const float* __restrict__ scales,
                         const float* __restrict__ mu1,
                         const float* __restrict__ mu2,
                         float* __restrict__ out)
{
    __shared__ _Float16 ldsA[2][BM * LDH];   // y = x*mu1, f16, [m][n]
    __shared__ _Float16 ldsW[2][BK * LDH];   // dequantized W, f16, [k][n]

    const int tid    = threadIdx.x;
    const int wave   = tid >> 5;
    const int lane   = tid & 31;
    const int lane16 = lane & 15;
    const int khalf  = lane >> 4;

    const int waveM  = wave & 1;             // 2 waves along batch
    const int waveK  = wave >> 1;            // 4 waves along output-K

    const int blockM = blockIdx.x * BM;
    const int blockK = blockIdx.y * BK;

    // Staging map: thread owns rows (r, r+64) and a 16-wide column quarter.
    const int q      = tid >> 6;             // 0..3
    const int r      = tid & 63;             // 0..63
    const int colOff = q * 16;

    const float* xbase = x  + (size_t)(blockM + r) * N_DIM + colOff;
    const int*   wbase = Wq + (size_t)(blockK + r) * N_DIM + colOff;
    const float* mu1c  = mu1 + colOff;
    const float* zrow0 = zeros  + (size_t)(blockK + r) * NGRP;
    const float* zrow1 = zeros  + (size_t)(blockK + r + 64) * NGRP;
    const float* srow0 = scales + (size_t)(blockK + r) * NGRP;
    const float* srow1 = scales + (size_t)(blockK + r + 64) * NGRP;
    const float  m2a   = mu2[blockK + r];
    const float  m2b   = mu2[blockK + r + 64];

    v8f acc[4][2];
    #pragma unroll
    for (int mt = 0; mt < 4; ++mt)
        #pragma unroll
        for (int kt = 0; kt < 2; ++kt)
            #pragma unroll
            for (int e = 0; e < 8; ++e)
                acc[mt][kt][e] = 0.0f;

    Stage stA, stB;
    load_stage(xbase, wbase, mu1c, zrow0, zrow1, srow0, srow1, m2a, m2b, 0, stA);

    // Software-pipelined, double-buffered main loop: one barrier per stage,
    // global loads for stage i+1 issued before the compute of stage i.
    for (int k0 = 0; k0 < N_DIM; k0 += 2 * SD) {
        deq_store(stA, ldsA[0], ldsW[0], r, colOff);
        load_stage(xbase, wbase, mu1c, zrow0, zrow1, srow0, srow1, m2a, m2b,
                   k0 + SD, stB);
        if (k0 + 2 * SD < N_DIM) {           // near-scope hint, 2 stages ahead
            __builtin_prefetch(xbase + k0 + 2 * SD, 0, 3);
            __builtin_prefetch(wbase + k0 + 2 * SD, 0, 3);
        }
        __syncthreads();
        compute_stage(ldsA[0], ldsW[0], waveM, waveK, lane16, khalf, acc);

        deq_store(stB, ldsA[1], ldsW[1], r, colOff);
        if (k0 + 2 * SD < N_DIM)
            load_stage(xbase, wbase, mu1c, zrow0, zrow1, srow0, srow1, m2a, m2b,
                       k0 + 2 * SD, stA);
        __syncthreads();
        compute_stage(ldsA[1], ldsW[1], waveM, waveK, lane16, khalf, acc);
    }

    // Epilogue: f32 C/D tile layout -> out[b,k]. VGPR e: row M = e + 8*khalf,
    // col N = lane16.
    #pragma unroll
    for (int mt = 0; mt < 4; ++mt) {
        #pragma unroll
        for (int kt = 0; kt < 2; ++kt) {
            const int col = blockK + waveK * 32 + kt * 16 + lane16;
            #pragma unroll
            for (int e = 0; e < 8; ++e) {
                const int row = blockM + waveM * 64 + mt * 16 + khalf * 8 + e;
                out[(size_t)row * K_DIM + col] = acc[mt][kt][e];
            }
        }
    }
}

extern "C" void kernel_launch(void* const* d_in, const int* in_sizes, int n_in,
                              void* d_out, int out_size, void* d_ws, size_t ws_size,
                              hipStream_t stream) {
    const float* x      = (const float*)d_in[0];
    const int*   Wq     = (const int*)  d_in[1];
    const float* zeros  = (const float*)d_in[2];
    const float* scales = (const float*)d_in[3];
    const float* mu1    = (const float*)d_in[4];
    const float* mu2    = (const float*)d_in[5];
    float* out = (float*)d_out;

    dim3 grid(B_DIM / BM, K_DIM / BK);       // 64 x 86 workgroups
    qlinear_wmma_kernel<<<grid, 256, 0, stream>>>(x, Wq, zeros, scales, mu1, mu2, out);
}
